// PerceiverAttentionBlock_62861141344528
// MI455X (gfx1250) — matrix operand order repaired and checked by reading, not compiled, not run on hardware
//
#include <hip/hip_runtime.h>

// ---------------------------------------------------------------------------
// CDNA5 (gfx1250) wave32 WMMA bf16 implementation of the Perceiver block.
// ---------------------------------------------------------------------------

typedef __attribute__((ext_vector_type(16))) __bf16 v16bf;
typedef __attribute__((ext_vector_type(8)))  float  v8f;

union FragU {
  v16bf v;
  uint4 u[2];
};

// fp32 -> bf16 via native convert (RNE), single v_cvt instead of ALU emulation.
__device__ __forceinline__ unsigned short f2bf(float f) {
  __bf16 h = (__bf16)f;
  return __builtin_bit_cast(unsigned short, h);
}

__device__ __forceinline__ uint4 pack8(float4 f0, float4 f1) {
  uint4 p;
  p.x = (unsigned)f2bf(f0.x) | ((unsigned)f2bf(f0.y) << 16);
  p.y = (unsigned)f2bf(f0.z) | ((unsigned)f2bf(f0.w) << 16);
  p.z = (unsigned)f2bf(f1.x) | ((unsigned)f2bf(f1.y) << 16);
  p.w = (unsigned)f2bf(f1.z) | ((unsigned)f2bf(f1.w) << 16);
  return p;
}

// A fragment (16x32 bf16, MxK) from row-major LDS [rows][ld].
// ISA layout: lanes 0-15 hold K {0..7,16..23}, lanes 16-31 hold K {8..15,24..31}.
__device__ __forceinline__ v16bf load_frag_A(const unsigned short* s, int row0,
                                             int k0, int ld, int lane) {
  const int m  = lane & 15;
  const int kb = (lane < 16) ? 0 : 8;
  const unsigned short* p = s + (row0 + m) * ld + k0 + kb;
  FragU f;
  f.u[0] = *(const uint4*)(p);        // K = kb .. kb+7
  f.u[1] = *(const uint4*)(p + 16);   // K = kb+16 .. kb+23
  return f.v;
}

// B fragment (32x16 bf16, KxN) where LDS holds row-major [N][K] (i.e. B = Wt).
// Lanes 0-15: N=lane, K {0..7,16..23}; lanes 16-31: N=lane-16, K {8..15,24..31}.
__device__ __forceinline__ v16bf load_frag_B(const unsigned short* s, int n0,
                                             int k0, int ld, int lane) {
  const int n  = lane & 15;
  const int kb = (lane < 16) ? 0 : 8;
  const unsigned short* p = s + (n0 + n) * ld + k0 + kb;
  FragU f;
  f.u[0] = *(const uint4*)(p);        // K = kb .. kb+7
  f.u[1] = *(const uint4*)(p + 16);   // K = kb+16 .. kb+23
  return f.v;
}

__device__ __forceinline__ v8f wmma_bf16(v16bf a, v16bf b, v8f c) {
  return __builtin_amdgcn_wmma_f32_16x16x32_bf16(
      false, a, false, b, (short)0, c, false, false);
}

// ---------------------------------------------------------------------------
// fp32 -> bf16 conversion (weights)
// ---------------------------------------------------------------------------
__global__ __launch_bounds__(256) void cvt_f32_bf16(const float* __restrict__ in,
                                                    unsigned short* __restrict__ out,
                                                    int n) {
  int i = blockIdx.x * 256 + threadIdx.x;
  if (i < n) out[i] = f2bf(in[i]);
}

// ---------------------------------------------------------------------------
// GEMM: Out[M][N](bf16) = scale * X[M][K](fp32) @ W[N][K](bf16)^T
// concat_mode=1: per-batch rows = 1088 = 1024 from X(k_v) + 64 from Xq(q).
// Block: 256 threads = 8 waves. Tile 128(M) x 128(N), BK=32, LDS double-buffer.
// Wave w: rows (w&3)*32, cols (w>>2)*64  ->  2x4 accum tiles, 8 WMMA / K-step.
// ---------------------------------------------------------------------------
__global__ __launch_bounds__(256) void gemm_f32_bf16(
    const float* __restrict__ X, const float* __restrict__ Xq,
    const unsigned short* __restrict__ Wb, unsigned short* __restrict__ Out,
    int M, int N, int K, float scale, int concat_mode) {
  __shared__ unsigned short As[2][128 * 40];
  __shared__ unsigned short Bs[2][128 * 40];

  const int tid = threadIdx.x;
  const int m0 = blockIdx.x * 128;
  const int n0 = blockIdx.y * 128;

  // Staging: thread -> row tid>>1 (0..127), col chunk (tid&1)*16.
  const int sr = tid >> 1;
  const int sc = (tid & 1) * 16;
  const float* arow;
  {
    int row_g = m0 + sr;
    if (concat_mode) {
      int batch = row_g / 1088;
      int s = row_g - batch * 1088;
      arow = (s < 1024)
                 ? (X + ((size_t)batch * 1024 + s) * (size_t)K)
                 : (Xq + ((size_t)batch * 64 + (s - 1024)) * (size_t)K);
    } else {
      arow = X + (size_t)row_g * (size_t)K;
    }
  }
  const unsigned short* brow = Wb + (size_t)(n0 + sr) * (size_t)K;

  const int wid = tid >> 5, lane = tid & 31;
  const int rg = (wid & 3) * 32;
  const int cg = (wid >> 2) * 64;

  const v8f vz = {0.f, 0.f, 0.f, 0.f, 0.f, 0.f, 0.f, 0.f};
  v8f acc[8] = {vz, vz, vz, vz, vz, vz, vz, vz};

  float4 fa0, fa1, fa2, fa3;
  uint4 fb0, fb1;

  // prologue: fetch k0=0 tile and park it in buffer 0
  fa0 = *(const float4*)(arow + sc);
  fa1 = *(const float4*)(arow + sc + 4);
  fa2 = *(const float4*)(arow + sc + 8);
  fa3 = *(const float4*)(arow + sc + 12);
  fb0 = ((const uint4*)(brow + sc))[0];
  fb1 = ((const uint4*)(brow + sc))[1];
  {
    uint4* da = (uint4*)&As[0][sr * 40 + sc];
    da[0] = pack8(fa0, fa1);
    da[1] = pack8(fa2, fa3);
    uint4* db = (uint4*)&Bs[0][sr * 40 + sc];
    db[0] = fb0;
    db[1] = fb1;
  }
  __syncthreads();

  int cur = 0;
  for (int k0 = 32; k0 <= K; k0 += 32) {
    const bool has_next = (k0 < K);
    if (has_next) {  // issue next global loads early (hidden behind WMMAs)
      fa0 = *(const float4*)(arow + k0 + sc);
      fa1 = *(const float4*)(arow + k0 + sc + 4);
      fa2 = *(const float4*)(arow + k0 + sc + 8);
      fa3 = *(const float4*)(arow + k0 + sc + 12);
      fb0 = ((const uint4*)(brow + k0 + sc))[0];
      fb1 = ((const uint4*)(brow + k0 + sc))[1];
    }
    // compute on current buffer: load all frags, then 8 back-to-back WMMAs
    v16bf a0 = load_frag_A(As[cur], rg, 0, 40, lane);
    v16bf a1 = load_frag_A(As[cur], rg + 16, 0, 40, lane);
    v16bf b0 = load_frag_B(Bs[cur], cg, 0, 40, lane);
    v16bf b1 = load_frag_B(Bs[cur], cg + 16, 0, 40, lane);
    v16bf b2 = load_frag_B(Bs[cur], cg + 32, 0, 40, lane);
    v16bf b3 = load_frag_B(Bs[cur], cg + 48, 0, 40, lane);
    acc[0] = wmma_bf16(a0, b0, acc[0]);
    acc[1] = wmma_bf16(a0, b1, acc[1]);
    acc[2] = wmma_bf16(a0, b2, acc[2]);
    acc[3] = wmma_bf16(a0, b3, acc[3]);
    acc[4] = wmma_bf16(a1, b0, acc[4]);
    acc[5] = wmma_bf16(a1, b1, acc[5]);
    acc[6] = wmma_bf16(a1, b2, acc[6]);
    acc[7] = wmma_bf16(a1, b3, acc[7]);
    if (has_next) {  // park next tile in the other buffer
      uint4* da = (uint4*)&As[cur ^ 1][sr * 40 + sc];
      da[0] = pack8(fa0, fa1);
      da[1] = pack8(fa2, fa3);
      uint4* db = (uint4*)&Bs[cur ^ 1][sr * 40 + sc];
      db[0] = fb0;
      db[1] = fb1;
    }
    __syncthreads();
    cur ^= 1;
  }

  const int hi = (lane >> 4) * 8;   // D frag: VGPR r holds M = r + hi
  const int nc = lane & 15;         // D frag: lane holds column N = nc
#pragma unroll
  for (int at = 0; at < 2; ++at) {
#pragma unroll
    for (int nt = 0; nt < 4; ++nt) {
#pragma unroll
      for (int r = 0; r < 8; ++r) {
        int m = m0 + rg + at * 16 + hi + r;
        int n = n0 + cg + nt * 16 + nc;
        Out[(size_t)m * (size_t)N + n] = f2bf(acc[at * 4 + nt][r] * scale);
      }
    }
  }
}

// ---------------------------------------------------------------------------
// GEMM: Out[M][N](fp32) = X[M][K](bf16) @ W[N][K](bf16)^T   (output projection)
// Same 128x128 double-buffered scheme; A is already bf16.
// ---------------------------------------------------------------------------
__global__ __launch_bounds__(256) void gemm_bf16_f32(
    const unsigned short* __restrict__ Xb, const unsigned short* __restrict__ Wb,
    float* __restrict__ Out, int M, int N, int K) {
  __shared__ unsigned short As[2][128 * 40];
  __shared__ unsigned short Bs[2][128 * 40];

  const int tid = threadIdx.x;
  const int m0 = blockIdx.x * 128;
  const int n0 = blockIdx.y * 128;

  const int sr = tid >> 1;
  const int sc = (tid & 1) * 16;
  const unsigned short* arow = Xb + (size_t)(m0 + sr) * (size_t)K;
  const unsigned short* brow = Wb + (size_t)(n0 + sr) * (size_t)K;

  const int wid = tid >> 5, lane = tid & 31;
  const int rg = (wid & 3) * 32;
  const int cg = (wid >> 2) * 64;

  const v8f vz = {0.f, 0.f, 0.f, 0.f, 0.f, 0.f, 0.f, 0.f};
  v8f acc[8] = {vz, vz, vz, vz, vz, vz, vz, vz};

  uint4 fa0, fa1, fb0, fb1;
  fa0 = ((const uint4*)(arow + sc))[0];
  fa1 = ((const uint4*)(arow + sc))[1];
  fb0 = ((const uint4*)(brow + sc))[0];
  fb1 = ((const uint4*)(brow + sc))[1];
  {
    uint4* da = (uint4*)&As[0][sr * 40 + sc];
    da[0] = fa0; da[1] = fa1;
    uint4* db = (uint4*)&Bs[0][sr * 40 + sc];
    db[0] = fb0; db[1] = fb1;
  }
  __syncthreads();

  int cur = 0;
  for (int k0 = 32; k0 <= K; k0 += 32) {
    const bool has_next = (k0 < K);
    if (has_next) {
      fa0 = ((const uint4*)(arow + k0 + sc))[0];
      fa1 = ((const uint4*)(arow + k0 + sc))[1];
      fb0 = ((const uint4*)(brow + k0 + sc))[0];
      fb1 = ((const uint4*)(brow + k0 + sc))[1];
    }
    v16bf a0 = load_frag_A(As[cur], rg, 0, 40, lane);
    v16bf a1 = load_frag_A(As[cur], rg + 16, 0, 40, lane);
    v16bf b0 = load_frag_B(Bs[cur], cg, 0, 40, lane);
    v16bf b1 = load_frag_B(Bs[cur], cg + 16, 0, 40, lane);
    v16bf b2 = load_frag_B(Bs[cur], cg + 32, 0, 40, lane);
    v16bf b3 = load_frag_B(Bs[cur], cg + 48, 0, 40, lane);
    acc[0] = wmma_bf16(a0, b0, acc[0]);
    acc[1] = wmma_bf16(a0, b1, acc[1]);
    acc[2] = wmma_bf16(a0, b2, acc[2]);
    acc[3] = wmma_bf16(a0, b3, acc[3]);
    acc[4] = wmma_bf16(a1, b0, acc[4]);
    acc[5] = wmma_bf16(a1, b1, acc[5]);
    acc[6] = wmma_bf16(a1, b2, acc[6]);
    acc[7] = wmma_bf16(a1, b3, acc[7]);
    if (has_next) {
      uint4* da = (uint4*)&As[cur ^ 1][sr * 40 + sc];
      da[0] = fa0; da[1] = fa1;
      uint4* db = (uint4*)&Bs[cur ^ 1][sr * 40 + sc];
      db[0] = fb0; db[1] = fb1;
    }
    __syncthreads();
    cur ^= 1;
  }

  const int hi = (lane >> 4) * 8;
  const int nc = lane & 15;
#pragma unroll
  for (int at = 0; at < 2; ++at) {
#pragma unroll
    for (int nt = 0; nt < 4; ++nt) {
#pragma unroll
      for (int r = 0; r < 8; ++r) {
        int m = m0 + rg + at * 16 + hi + r;
        int n = n0 + cg + nt * 16 + nc;
        Out[(size_t)m * (size_t)N + n] = acc[at * 4 + nt][r];
      }
    }
  }
}

// ---------------------------------------------------------------------------
// Attention: 1 block per (batch, head). 128 threads = 4 waves.
// Wave w owns q columns [16w, 16w+16). Computes scores TRANSPOSED
// (S^T = K @ q^T) so softmax over s stays per-lane (one shfl_xor across lane
// halves). Online (flash) softmax; O accumulated transposed: o^T = V^T @ P^T.
// qp is pre-scaled by sqrt(d)=8.
// ---------------------------------------------------------------------------
__global__ __launch_bounds__(128) void attn_kernel(
    const unsigned short* __restrict__ qp,    // [32][64][512] bf16
    const unsigned short* __restrict__ kvp,   // [32][1088][1024] bf16 (k|v)
    unsigned short* __restrict__ obuf) {      // [32][64][512] bf16
  __shared__ unsigned short Qs[64 * 72];    // [lq][d]
  __shared__ unsigned short Ks[32 * 72];    // [s_chunk][d]
  __shared__ unsigned short VTs[64 * 40];   // [d][s_chunk]  (transposed V)
  __shared__ unsigned short Ps[64 * 40];    // [lq][s_chunk] softmaxed probs

  const int tid = threadIdx.x;
  const int bh = blockIdx.x;
  const int batch = bh >> 3;
  const int head = bh & 7;

  {  // stage Q (64x64 bf16)
    int r = tid >> 1;
    int c = (tid & 1) * 32;
    const unsigned short* src =
        qp + ((size_t)(batch * 64 + r) * 512) + head * 64 + c;
    const uint4* s4 = (const uint4*)src;
    uint4* dst = (uint4*)&Qs[r * 72 + c];
    dst[0] = s4[0]; dst[1] = s4[1]; dst[2] = s4[2]; dst[3] = s4[3];
  }
  __syncthreads();

  const int wid = tid >> 5, lane = tid & 31;
  const int qc = wid * 16;

  // Loop-invariant q fragments (B operand of scores^T)
  v16bf bq0 = load_frag_B(Qs, qc, 0, 72, lane);
  v16bf bq1 = load_frag_B(Qs, qc, 32, 72, lane);

  const v8f vz = {0.f, 0.f, 0.f, 0.f, 0.f, 0.f, 0.f, 0.f};
  v8f oacc[4] = {vz, vz, vz, vz};
  float m_run = -3.0e38f, l_run = 0.f;

  const int hi = (lane >> 4) * 8;
  const int qrow = qc + (lane & 15);

  for (int s0 = 0; s0 < 1088; s0 += 32) {
    {  // stage K chunk [32][64]
      int si = tid >> 2;
      int c = (tid & 3) * 16;
      const unsigned short* src =
          kvp + ((size_t)(batch * 1088 + s0 + si) * 1024) + head * 64 + c;
      const uint4* s4 = (const uint4*)src;
      uint4* dst = (uint4*)&Ks[si * 72 + c];
      dst[0] = s4[0]; dst[1] = s4[1];
    }
    {  // stage V chunk transposed -> VTs[d][si]
      int si = tid >> 2;
      int c = (tid & 3) * 16;
      const unsigned short* src =
          kvp + ((size_t)(batch * 1088 + s0 + si) * 1024) + 512 + head * 64 + c;
      unsigned short e[16];
      *(uint4*)&e[0] = ((const uint4*)src)[0];
      *(uint4*)&e[8] = ((const uint4*)src)[1];
#pragma unroll
      for (int j = 0; j < 16; ++j) VTs[(c + j) * 40 + si] = e[j];
    }
    __syncthreads();

    // scores^T tiles: D[s][q], lane holds q col = qrow, rows s = hi+r (+16)
    v16bf a00 = load_frag_A(Ks, 0, 0, 72, lane);
    v16bf a01 = load_frag_A(Ks, 0, 32, 72, lane);
    v16bf a10 = load_frag_A(Ks, 16, 0, 72, lane);
    v16bf a11 = load_frag_A(Ks, 16, 32, 72, lane);
    v8f d0 = wmma_bf16(a00, bq0, vz);
    v8f d1 = wmma_bf16(a10, bq0, vz);
    d0 = wmma_bf16(a01, bq1, d0);
    d1 = wmma_bf16(a11, bq1, d1);

    // online softmax along s (per q column; lane pair L / L+16 shares column)
    float cmax = -3.0e38f;
#pragma unroll
    for (int r = 0; r < 8; ++r) cmax = fmaxf(cmax, fmaxf(d0[r], d1[r]));
    cmax = fmaxf(cmax, __shfl_xor(cmax, 16, 32));
    float m_new = fmaxf(m_run, cmax);
    float corr = __expf(m_run - m_new);
    float p0[8], p1[8];
    float psum = 0.f;
#pragma unroll
    for (int r = 0; r < 8; ++r) {
      p0[r] = __expf(d0[r] - m_new);
      p1[r] = __expf(d1[r] - m_new);
      psum += p0[r] + p1[r];
    }
    psum += __shfl_xor(psum, 16, 32);
    l_run = l_run * corr + psum;
    m_run = m_new;

#pragma unroll
    for (int t = 0; t < 4; ++t)
#pragma unroll
      for (int r = 0; r < 8; ++r) oacc[t][r] *= corr;

    // write P transposed-back: Ps[q][s] (each wave writes only its own rows)
#pragma unroll
    for (int r = 0; r < 8; ++r) {
      Ps[qrow * 40 + hi + r] = f2bf(p0[r]);
      Ps[qrow * 40 + 16 + hi + r] = f2bf(p1[r]);
    }

    // o^T += V^T @ P^T : A = VTs[d][s], B = Ps rows (this wave's own rows)
    v16bf pb = load_frag_B(Ps, qc, 0, 40, lane);
    v16bf av0 = load_frag_A(VTs, 0, 0, 40, lane);
    v16bf av1 = load_frag_A(VTs, 16, 0, 40, lane);
    v16bf av2 = load_frag_A(VTs, 32, 0, 40, lane);
    v16bf av3 = load_frag_A(VTs, 48, 0, 40, lane);
    oacc[0] = wmma_bf16(av0, pb, oacc[0]);
    oacc[1] = wmma_bf16(av1, pb, oacc[1]);
    oacc[2] = wmma_bf16(av2, pb, oacc[2]);
    oacc[3] = wmma_bf16(av3, pb, oacc[3]);
    __syncthreads();
  }

  const float inv_l = 1.0f / l_run;
#pragma unroll
  for (int dt = 0; dt < 4; ++dt) {
#pragma unroll
    for (int r = 0; r < 8; ++r) {
      int d = dt * 16 + hi + r;
      obuf[((size_t)(batch * 64 + qrow)) * 512 + head * 64 + d] =
          f2bf(oacc[dt][r] * inv_l);
    }
  }
}

// ---------------------------------------------------------------------------
// Launch
// Inputs: q [32,64,1024] f32 | k_v [32,1024,1024] f32 | W_kv [1024,1024] f32
//         W_q [512,1024] f32 | W_out [1024,512] f32.  Output [32,64,1024] f32.
// ---------------------------------------------------------------------------
extern "C" void kernel_launch(void* const* d_in, const int* in_sizes, int n_in,
                              void* d_out, int out_size, void* d_ws,
                              size_t ws_size, hipStream_t stream) {
  (void)in_sizes; (void)n_in; (void)out_size; (void)ws_size;
  const float* q = (const float*)d_in[0];
  const float* k_v = (const float*)d_in[1];
  const float* Wkv = (const float*)d_in[2];
  const float* Wq = (const float*)d_in[3];
  const float* Wout = (const float*)d_in[4];
  float* out = (float*)d_out;

  unsigned short* ws = (unsigned short*)d_ws;
  unsigned short* wkv_b  = ws;                 // 1,048,576
  unsigned short* wq_b   = ws + 1048576;       //   524,288
  unsigned short* wout_b = ws + 1572864;       //   524,288
  unsigned short* qp_b   = ws + 2097152;       // 1,048,576 (q proj, *8)
  unsigned short* o_b    = ws + 3145728;       // 1,048,576 (attn out)
  unsigned short* kvp_b  = ws + 4194304;       // 35,651,584 (kv proj)

  // 0) weights -> bf16
  cvt_f32_bf16<<<(1048576 + 255) / 256, 256, 0, stream>>>(Wkv, wkv_b, 1048576);
  cvt_f32_bf16<<<(524288 + 255) / 256, 256, 0, stream>>>(Wq, wq_b, 524288);
  cvt_f32_bf16<<<(524288 + 255) / 256, 256, 0, stream>>>(Wout, wout_b, 524288);

  // 1) qp = (q @ W_q^T) * sqrt(64)   [2048 x 512]
  {
    dim3 grid(2048 / 128, 512 / 128);
    gemm_f32_bf16<<<grid, 256, 0, stream>>>(q, nullptr, wq_b, qp_b, 2048, 512,
                                            1024, 8.0f, 0);
  }
  // 2) kvp = concat(k_v, q) @ W_kv^T   [34816 x 1024], concat fused into rows
  {
    dim3 grid(34816 / 128, 1024 / 128);
    gemm_f32_bf16<<<grid, 256, 0, stream>>>(k_v, q, wkv_b, kvp_b, 34816, 1024,
                                            1024, 1.0f, 1);
  }
  // 3) flash attention per (batch, head)
  attn_kernel<<<256, 128, 0, stream>>>(qp_b, kvp_b, o_b);

  // 4) out = o @ W_out^T  [2048 x 1024] fp32
  {
    dim3 grid(2048 / 128, 1024 / 128);
    gemm_bf16_f32<<<grid, 256, 0, stream>>>(o_b, wout_b, out, 2048, 1024, 512);
  }
}
